// RelativePositionalEncoder_33689723469912
// MI455X (gfx1250) — matrix operand order repaired and checked by reading, not compiled, not run on hardware
//
#include <hip/hip_runtime.h>
#include <stdint.h>

typedef __attribute__((ext_vector_type(16))) __bf16 v16bf;
typedef __attribute__((ext_vector_type(8)))  float  v8f;

constexpr int SEQ   = 512;
constexpr int DM    = 512;
constexpr int BATCH = 8;
constexpr int KTOT  = 1024;   // 512 (Q·K^T part) + 512 (W·E part)

constexpr int BM = 128;       // fallback block tile M
constexpr int BN = 64;        // fallback block tile N
constexpr int BK = 32;        // k-step (bf16 wmma K)
constexpr int THREADS = 256;  // fallback: 8 wave32

constexpr int BMF = 128;      // fast-path block tile M
constexpr int BNF = 128;      // fast-path block tile N
constexpr int THREADS_F = 512;// fast-path: 16 wave32, wave tile 32x32
constexpr int NKS = KTOT / BK;

// ---------- split fp32 x ~= hi + lo (both bf16, RNE), bitwise ----------
__device__ __forceinline__ void split1(float x, uint16_t& hb, uint16_t& lb) {
  uint32_t u = __builtin_bit_cast(uint32_t, x);
  uint32_t r = u + 0x7fffu + ((u >> 16) & 1u);
  hb = (uint16_t)(r >> 16);
  float hf = __builtin_bit_cast(float, (uint32_t)hb << 16);
  float lo = x - hf;
  uint32_t ul = __builtin_bit_cast(uint32_t, lo);
  uint32_t rl = ul + 0x7fffu + ((ul >> 16) & 1u);
  lb = (uint16_t)(rl >> 16);
}

__device__ __forceinline__ uint32_t pack_hi_lo(float x) {  // {lo[31:16], hi[15:0]}
  uint16_t hb, lb; split1(x, hb, lb);
  return ((uint32_t)lb << 16) | (uint32_t)hb;
}

// ---------- Pass 1: Wcol0[b*SEQ+i] = q[b,i,i] + sum_{j>i} q[b,i,j] ----------
__global__ void relpos_wcol0_kernel(const float* __restrict__ q,
                                    float* __restrict__ wcol0) {
  int row  = blockIdx.x * 8 + (threadIdx.x >> 5);
  int lane = threadIdx.x & 31;
  int b = row >> 9;
  int i = row & 511;
  const float* qr = q + ((size_t)b * SEQ + i) * DM;
  float s = 0.f;
  for (int j = i + 1 + lane; j < DM; j += 32) s += qr[j];
  #pragma unroll
  for (int off = 16; off > 0; off >>= 1) s += __shfl_xor(s, off, 32);
  if (lane == 0) wcol0[row] = s + qr[i];
}

// ---------- Pass 2a: split fp32 array -> bf16 hi/lo planes (float4 wide) ----------
__global__ void relpos_split4_kernel(const float* __restrict__ src,
                                     uint16_t* __restrict__ hi,
                                     uint16_t* __restrict__ lo, int n4) {
  int g = blockIdx.x * blockDim.x + threadIdx.x;
  if (g >= n4) return;
  float4 v = ((const float4*)src)[g];
  uint16_t h0, h1, h2, h3, l0, l1, l2, l3;
  split1(v.x, h0, l0); split1(v.y, h1, l1);
  split1(v.z, h2, l2); split1(v.w, h3, l3);
  uint2 hp = make_uint2((uint32_t)h0 | ((uint32_t)h1 << 16),
                        (uint32_t)h2 | ((uint32_t)h3 << 16));
  uint2 lp = make_uint2((uint32_t)l0 | ((uint32_t)l1 << 16),
                        (uint32_t)l2 | ((uint32_t)l3 << 16));
  ((uint2*)hi)[g] = hp;
  ((uint2*)lo)[g] = lp;
}

// ---------- Pass 2b: W[b,i,t] gathered from Q (+ clip column), split ----------
__global__ void relpos_buildw_kernel(const float* __restrict__ q,
                                     const float* __restrict__ wcol0,
                                     uint16_t* __restrict__ whi,
                                     uint16_t* __restrict__ wlo) {
  int idx = blockIdx.x * 256 + threadIdx.x;     // b*2^18 + i*2^9 + t
  int t = idx & (DM - 1);
  int i = (idx >> 9) & (SEQ - 1);
  int b = idx >> 18;
  float v = 0.f;
  if (t == 0)      v = wcol0[b * SEQ + i];
  else if (t <= i) v = q[((size_t)b * SEQ + i) * DM + (i - t)];
  uint16_t hb, lb; split1(v, hb, lb);
  whi[idx] = hb; wlo[idx] = lb;
}

// ---------- Pass 2c: Et[n][t] = E[t][n], split ----------
__global__ void relpos_buildet_kernel(const float* __restrict__ emb,
                                      uint16_t* __restrict__ ethi,
                                      uint16_t* __restrict__ etlo) {
  int idx = blockIdx.x * 256 + threadIdx.x;     // n*512 + t
  int t = idx & (DM - 1);
  int n = idx >> 9;
  float v = emb[(size_t)t * DM + n];
  uint16_t hb, lb; split1(v, hb, lb);
  ethi[idx] = hb; etlo[idx] = lb;
}

// ---------- Pass 3: out[b] = [Q|W] x [K^T;E], split-bf16 WMMA ----------
// Double-buffered LDS, tiles pre-swizzled into WMMA fragment layout:
//   l*[buf][sub][lane 0..31][slot 0..15]  slots 0..7 = hi (16 bf16), 8..15 = lo
//   A elem (M=r,K=c):  lane=(r&15)+16*((c>>3)&1), e=(c&7)|((c>>4)<<3)
//   B elem (K=kk,N=n): lane=(n&15)+16*((kk>>3)&1), e=(kk&7)|((kk>>4)<<3)
// Block tile 128x128, 16 wave32, wave tile 32x32 (2x2 WMMA subtiles).
// Peak VGPR pressure ~112/lane: 32 acc + 32 A-frag + 16 B-frag + 16 staging.
__global__ __launch_bounds__(THREADS_F)
void relpos_gemm_fast_kernel(const uint16_t* __restrict__ Qhi, const uint16_t* __restrict__ Qlo,
                             const uint16_t* __restrict__ Khi, const uint16_t* __restrict__ Klo,
                             const uint16_t* __restrict__ Whi, const uint16_t* __restrict__ Wlo,
                             const uint16_t* __restrict__ Ethi, const uint16_t* __restrict__ Etlo,
                             float* __restrict__ out) {
  __shared__ alignas(64) uint32_t lA[2][8][32][16];  // 32 KB
  __shared__ alignas(64) uint32_t lB[2][8][32][16];  // 32 KB

  const int b    = blockIdx.z;
  const int i0   = blockIdx.y * BMF;
  const int n0   = blockIdx.x * BNF;
  const int tid  = threadIdx.x;
  const int lane = tid & 31;
  const int wave = tid >> 5;                 // 0..15: waveM = wave&3, waveN = wave>>2
  const int hh   = lane >> 4;
  const int lm   = lane & 15;
  const uint32_t bOff = (uint32_t)b * SEQ * DM;

  // staging ownership: one A group + one B group per thread (row gr, 8 cols at gc)
  const int gr = tid >> 2, gc = (tid & 3) * 8;   // gr 0..127, gc {0,8,16,24}
  const int sS    = gr >> 4;
  const int laneA = (gr & 15) + 16 * ((gc >> 3) & 1);
  const int slotA = (gc >> 4) * 4;
  const int laneB = (gr & 15) + 16 * ((gc >> 3) & 1);
  const int slotB = (gc >> 4) * 4;
  // per-thread 32-bit offsets (SGPR base + VGPR offset addressing)
  const uint32_t offA = (uint32_t)(i0 + gr) * DM + gc;
  const uint32_t offB = (uint32_t)(n0 + gr) * DM + gc;

  uint4 aH, aL, bH, bL;
  auto load_tile = [&](int k0) {
    const uint16_t *Ah, *Al, *Bh, *Bl; uint32_t cc;
    if (k0 < DM) { Ah = Qhi + bOff; Al = Qlo + bOff; Bh = Khi + bOff; Bl = Klo + bOff; cc = (uint32_t)k0; }
    else         { Ah = Whi + bOff; Al = Wlo + bOff; Bh = Ethi;       Bl = Etlo;       cc = (uint32_t)(k0 - DM); }
    aH = *(const uint4*)(Ah + (offA + cc));
    aL = *(const uint4*)(Al + (offA + cc));
    bH = *(const uint4*)(Bh + (offB + cc));
    bL = *(const uint4*)(Bl + (offB + cc));
  };
  auto store_tile = [&](int buf) {
    uint32_t* pa = &lA[buf][sS][laneA][slotA];
    uint32_t* pb = &lB[buf][sS][laneB][slotB];
    *(uint4*)pa       = aH;
    *(uint4*)(pa + 8) = aL;
    *(uint4*)pb       = bH;
    *(uint4*)(pb + 8) = bL;
  };

  v8f acc[2][2];
  #pragma unroll
  for (int mi = 0; mi < 2; ++mi)
    #pragma unroll
    for (int ni = 0; ni < 2; ++ni)
      acc[mi][ni] = (v8f){0.f, 0.f, 0.f, 0.f, 0.f, 0.f, 0.f, 0.f};

  load_tile(0);
  store_tile(0);

  for (int ks = 0; ks < NKS; ++ks) {
    __syncthreads();               // buf[cur] visible to all waves
    const int cur = ks & 1;
    if (ks + 1 < NKS) load_tile((ks + 1) * BK);  // overlap global loads w/ WMMA

    // A fragments held across the K-step (32 VGPRs)
    v16bf fAh[2], fAl[2];
    #pragma unroll
    for (int mi = 0; mi < 2; ++mi) {
      const uint32_t* p = &lA[cur][(wave & 3) * 2 + mi][lane][0];
      fAh[mi] = *(const v16bf*)p;
      fAl[mi] = *(const v16bf*)(p + 8);
    }
    // B fragments streamed per-ni (16 VGPRs live at a time)
    #pragma unroll
    for (int ni = 0; ni < 2; ++ni) {
      const uint32_t* p = &lB[cur][(wave >> 2) * 2 + ni][lane][0];
      v16bf fBh = *(const v16bf*)p;
      v16bf fBl = *(const v16bf*)(p + 8);
      #pragma unroll
      for (int mi = 0; mi < 2; ++mi) {
        acc[mi][ni] = __builtin_amdgcn_wmma_f32_16x16x32_bf16(
            false, fAh[mi], false, fBh, (short)0, acc[mi][ni], false, false);
        acc[mi][ni] = __builtin_amdgcn_wmma_f32_16x16x32_bf16(
            false, fAh[mi], false, fBl, (short)0, acc[mi][ni], false, false);
        acc[mi][ni] = __builtin_amdgcn_wmma_f32_16x16x32_bf16(
            false, fAl[mi], false, fBh, (short)0, acc[mi][ni], false, false);
      }
    }

    if (ks + 1 < NKS) store_tile(cur ^ 1);  // safe: nobody reads buf^1 this epoch
  }

  float* ob = out + (size_t)b * SEQ * SEQ;
  #pragma unroll
  for (int mi = 0; mi < 2; ++mi)
    #pragma unroll
    for (int ni = 0; ni < 2; ++ni) {
      uint32_t n = (uint32_t)(n0 + (wave >> 2) * 32 + ni * 16 + lm);
      #pragma unroll
      for (int r = 0; r < 8; ++r) {
        uint32_t m = (uint32_t)(i0 + (wave & 3) * 32 + mi * 16 + r + hh * 8);
        ob[m * (uint32_t)SEQ + n] = acc[mi][ni][r];
      }
    }
}

// ---------- Fallback (small workspace): fused kernel ----------
__global__ __launch_bounds__(THREADS)
void relpos_gemm_fused_kernel(const float* __restrict__ q,
                              const float* __restrict__ kmat,
                              const float* __restrict__ emb,
                              const float* __restrict__ wcol0,
                              float* __restrict__ out) {
  __shared__ uint32_t lA[BM][BK + 1];
  __shared__ uint32_t lB[BK][BN + 1];

  const int b    = blockIdx.z;
  const int i0   = blockIdx.y * BM;
  const int n0   = blockIdx.x * BN;
  const int tid  = threadIdx.x;
  const int lane = tid & 31;
  const int wave = tid >> 5;
  const int waveM = (wave & 3) * 32;
  const int waveN = (wave >> 2) * 32;
  const int hh = lane >> 4;
  const int lm = lane & 15;

  const float* qb = q    + (size_t)b * SEQ * DM;
  const float* kb = kmat + (size_t)b * SEQ * DM;

  v8f acc[2][2];
  #pragma unroll
  for (int mi = 0; mi < 2; ++mi)
    #pragma unroll
    for (int ni = 0; ni < 2; ++ni)
      acc[mi][ni] = (v8f){0.f, 0.f, 0.f, 0.f, 0.f, 0.f, 0.f, 0.f};

  for (int k0 = 0; k0 < KTOT; k0 += BK) {
    if (k0 < DM) {
      for (int idx = tid; idx < BM * BK; idx += THREADS) {
        int r = idx >> 5, c = idx & 31;
        lA[r][c] = pack_hi_lo(qb[(size_t)(i0 + r) * DM + (k0 + c)]);
      }
    } else {
      int t0 = k0 - DM;
      for (int idx = tid; idx < BM * BK; idx += THREADS) {
        int r = idx >> 5, c = idx & 31;
        int i = i0 + r, t = t0 + c;
        float v;
        if (t == 0)      v = wcol0[b * SEQ + i];
        else if (t <= i) v = qb[(size_t)i * DM + (i - t)];
        else             v = 0.f;
        lA[r][c] = pack_hi_lo(v);
      }
    }
    if (k0 < DM) {
      for (int idx = tid; idx < BK * BN; idx += THREADS) {
        int n = idx >> 5, kk = idx & 31;
        lB[kk][n] = pack_hi_lo(kb[(size_t)(n0 + n) * DM + (k0 + kk)]);
      }
    } else {
      int t0 = k0 - DM;
      for (int idx = tid; idx < BK * BN; idx += THREADS) {
        int kk = idx >> 6, n = idx & 63;
        lB[kk][n] = pack_hi_lo(emb[(size_t)(t0 + kk) * DM + (n0 + n)]);
      }
    }
    __syncthreads();

    v16bf aHi[2], aLo[2], bHi[2], bLo[2];
    #pragma unroll
    for (int mi = 0; mi < 2; ++mi) {
      int m = waveM + mi * 16 + lm;
      #pragma unroll
      for (int e = 0; e < 16; ++e) {
        int kk = (e & 7) + ((e >> 3) << 4) + (hh << 3);
        uint32_t p = lA[m][kk];
        aHi[mi][e] = __builtin_bit_cast(__bf16, (uint16_t)(p & 0xffffu));
        aLo[mi][e] = __builtin_bit_cast(__bf16, (uint16_t)(p >> 16));
      }
    }
    #pragma unroll
    for (int ni = 0; ni < 2; ++ni) {
      int n = waveN + ni * 16 + lm;
      #pragma unroll
      for (int e = 0; e < 16; ++e) {
        int kk = (e & 7) + ((e >> 3) << 4) + (hh << 3);
        uint32_t p = lB[kk][n];
        bHi[ni][e] = __builtin_bit_cast(__bf16, (uint16_t)(p & 0xffffu));
        bLo[ni][e] = __builtin_bit_cast(__bf16, (uint16_t)(p >> 16));
      }
    }
    #pragma unroll
    for (int mi = 0; mi < 2; ++mi)
      #pragma unroll
      for (int ni = 0; ni < 2; ++ni) {
        acc[mi][ni] = __builtin_amdgcn_wmma_f32_16x16x32_bf16(
            false, aHi[mi], false, bHi[ni], (short)0, acc[mi][ni], false, false);
        acc[mi][ni] = __builtin_amdgcn_wmma_f32_16x16x32_bf16(
            false, aHi[mi], false, bLo[ni], (short)0, acc[mi][ni], false, false);
        acc[mi][ni] = __builtin_amdgcn_wmma_f32_16x16x32_bf16(
            false, aLo[mi], false, bHi[ni], (short)0, acc[mi][ni], false, false);
      }
    __syncthreads();
  }

  float* ob = out + (size_t)b * SEQ * SEQ;
  #pragma unroll
  for (int mi = 0; mi < 2; ++mi)
    #pragma unroll
    for (int ni = 0; ni < 2; ++ni) {
      int n = n0 + waveN + ni * 16 + lm;
      #pragma unroll
      for (int r = 0; r < 8; ++r) {
        int m = i0 + waveM + mi * 16 + r + hh * 8;
        ob[(size_t)m * SEQ + n] = acc[mi][ni][r];
      }
    }
}

extern "C" void kernel_launch(void* const* d_in, const int* in_sizes, int n_in,
                              void* d_out, int out_size, void* d_ws, size_t ws_size,
                              hipStream_t stream) {
  const float* q   = (const float*)d_in[0];
  const float* k   = (const float*)d_in[1];
  const float* emb = (const float*)d_in[2];
  float* out = (float*)d_out;

  constexpr size_t NQK   = (size_t)BATCH * SEQ * DM;      // 2M elements
  constexpr size_t PLANE = NQK * sizeof(uint16_t);        // 4 MB
  constexpr size_t EPL   = (size_t)SEQ * DM * sizeof(uint16_t);  // 0.5 MB
  constexpr size_t OFF_W0  = 0;                           // 16 KB
  constexpr size_t OFF_QHI = 16 * 1024;
  constexpr size_t OFF_QLO = OFF_QHI + PLANE;
  constexpr size_t OFF_KHI = OFF_QLO + PLANE;
  constexpr size_t OFF_KLO = OFF_KHI + PLANE;
  constexpr size_t OFF_WHI = OFF_KLO + PLANE;
  constexpr size_t OFF_WLO = OFF_WHI + PLANE;
  constexpr size_t OFF_EHI = OFF_WLO + PLANE;
  constexpr size_t OFF_ELO = OFF_EHI + EPL;
  constexpr size_t NEEDED  = OFF_ELO + EPL;               // ~25 MB

  float* wcol0 = (float*)((char*)d_ws + OFF_W0);
  relpos_wcol0_kernel<<<(BATCH * SEQ) / 8, 256, 0, stream>>>(q, wcol0);

  if (ws_size >= NEEDED) {
    uint16_t* Qhi  = (uint16_t*)((char*)d_ws + OFF_QHI);
    uint16_t* Qlo  = (uint16_t*)((char*)d_ws + OFF_QLO);
    uint16_t* Khi  = (uint16_t*)((char*)d_ws + OFF_KHI);
    uint16_t* Klo  = (uint16_t*)((char*)d_ws + OFF_KLO);
    uint16_t* Whi  = (uint16_t*)((char*)d_ws + OFF_WHI);
    uint16_t* Wlo  = (uint16_t*)((char*)d_ws + OFF_WLO);
    uint16_t* Ethi = (uint16_t*)((char*)d_ws + OFF_EHI);
    uint16_t* Etlo = (uint16_t*)((char*)d_ws + OFF_ELO);

    relpos_split4_kernel<<<(int)(NQK / 4 / 256), 256, 0, stream>>>(q, Qhi, Qlo, (int)(NQK / 4));
    relpos_split4_kernel<<<(int)(NQK / 4 / 256), 256, 0, stream>>>(k, Khi, Klo, (int)(NQK / 4));
    relpos_buildw_kernel<<<(int)(NQK / 256), 256, 0, stream>>>(q, wcol0, Whi, Wlo);
    relpos_buildet_kernel<<<(SEQ * DM) / 256, 256, 0, stream>>>(emb, Ethi, Etlo);

    dim3 gridF(SEQ / BNF, SEQ / BMF, BATCH);  // 4 x 4 x 8
    relpos_gemm_fast_kernel<<<gridF, THREADS_F, 0, stream>>>(
        Qhi, Qlo, Khi, Klo, Whi, Wlo, Ethi, Etlo, out);
  } else {
    dim3 grid(SEQ / BN, SEQ / BM, BATCH);     // 8 x 4 x 8
    relpos_gemm_fused_kernel<<<grid, THREADS, 0, stream>>>(q, k, emb, wcol0, out);
  }
}
